// SimpleModelWithNNEnding_83580063580829
// MI455X (gfx1250) — matrix operand order repaired and checked by reading, not compiled, not run on hardware
//
#include <hip/hip_runtime.h>
#include <hip/hip_bf16.h>

typedef __attribute__((ext_vector_type(16))) _Float16 v16h;
typedef __attribute__((ext_vector_type(8)))  float    v8f;

#define NN 100000
#define NE 3200000
#define LDW 68   // padded LDS row stride (floats) to avoid bank conflicts

#if __has_builtin(__builtin_amdgcn_wave_barrier)
#define WAVE_SYNC() __builtin_amdgcn_wave_barrier()
#else
#define WAVE_SYNC() __syncthreads()
#endif

#define WMMA(a, b, c) __builtin_amdgcn_wmma_f32_16x16x32_f16(false, (a), false, (b), (short)0, (c), false, false)

// ---------- monotonic float<->uint mapping for atomicMax-based segment max ----------
__device__ __forceinline__ unsigned enc_ord(float f) {
  unsigned u = __float_as_uint(f);
  return (u & 0x80000000u) ? ~u : (u | 0x80000000u);
}
__device__ __forceinline__ float dec_ord(unsigned e) {
  unsigned u = (e & 0x80000000u) ? (e & 0x7FFFFFFFu) : ~e;
  return __uint_as_float(u);   // enc==0 (untouched) decodes to NaN -> treated as "no edges"
}

// ---------- WMMA fragment helpers (layouts per CDNA5 ISA §7.12.2) ----------
// A (16x32 f16): lane L -> row M=L&15; elems i<8: K = base+i, i>=8: K = 16+base+(i-8), base=(L>=16)?8:0
__device__ __forceinline__ v16h loadA(const float (*buf)[LDW], int lane, int kstep) {
  int row = lane & 15;
  int kb  = kstep * 32 + ((lane >> 4) << 3);
  const float* r = buf[row];
  float4 p0 = *(const float4*)(r + kb);
  float4 p1 = *(const float4*)(r + kb + 4);
  float4 p2 = *(const float4*)(r + kb + 16);
  float4 p3 = *(const float4*)(r + kb + 20);
  v16h a;
  a[0]=(_Float16)p0.x;  a[1]=(_Float16)p0.y;  a[2]=(_Float16)p0.z;  a[3]=(_Float16)p0.w;
  a[4]=(_Float16)p1.x;  a[5]=(_Float16)p1.y;  a[6]=(_Float16)p1.z;  a[7]=(_Float16)p1.w;
  a[8]=(_Float16)p2.x;  a[9]=(_Float16)p2.y;  a[10]=(_Float16)p2.z; a[11]=(_Float16)p2.w;
  a[12]=(_Float16)p3.x; a[13]=(_Float16)p3.y; a[14]=(_Float16)p3.z; a[15]=(_Float16)p3.w;
  return a;
}
// B tiles pre-packed per-lane-contiguous in workspace: 512 f16 per tile, lane gets 16 contiguous f16.
__device__ __forceinline__ v16h loadB(const _Float16* wB, int tile, int lane) {
  return *(const v16h*)(wB + tile * 512 + lane * 16);
}
// D/C (16x16 f32): lane L -> col N=L&15; VGPR r -> row M = r + (L>>4)*8
__device__ __forceinline__ void storeD(float (*buf)[LDW], int lane, v8f acc,
                                       int nc, const float* bias, bool dorelu) {
  int col = nc * 16 + (lane & 15);
  int rb  = (lane >> 4) * 8;
  float b = bias[col];
#pragma unroll
  for (int r = 0; r < 8; ++r) {
    float v = acc[r] + b;
    if (dorelu) v = fmaxf(v, 0.f);
    buf[rb + r][col] = v;
  }
}

// ---------- Kernel 0: pack f16 B-tiles + fold attention edge vectors ----------
__global__ void k_prep(const float* c1We, const float* c1ae,
                       const float* c2We, const float* c2ae,
                       const float* w_1, const float* w_2, const float* w_3, const float* w_4,
                       const float* hw1, const float* hw2,
                       float* wvec, _Float16* wB) {
  int t = threadIdx.x;
  if (t < 6)  { float s = 0.f; for (int j = 0; j < 16; ++j) s += c1ae[j] * c1We[j * 6 + t];  wvec[t] = s; }
  if (t >= 8 && t < 24) {
    int k = t - 8; float s = 0.f;
    for (int j = 0; j < 16; ++j) s += c2ae[j] * c2We[j * 16 + k];
    wvec[8 + k] = s;
  }
  const int tl[15] = {0,0,0,0, 1,1,1,1, 2,2,2,2, 3, 4, 5};
  const int tk[15] = {0,0,1,1, 0,0,0,0, 0,0,1,1, 0, 0, 0};
  const int tn[15] = {0,1,0,1, 0,1,2,3, 0,1,0,1, 0, 0, 0};
  const int Kin[6]  = {38,32,64,32,16,16};
  const int Nout[6] = {32,64,32,16,16,4};
  for (int idx = threadIdx.x; idx < 15 * 512; idx += blockDim.x) {
    int tile = idx >> 9, lane = (idx >> 4) & 31, i = idx & 15;
    int L = tl[tile];
    int K = tk[tile] * 32 + ((i >> 3) << 4) + ((lane >> 4) << 3) + (i & 7);   // B layout: lane=N, K=(elem>=8)*16+halfsel*8+(elem&7)
    int N = tn[tile] * 16 + (lane & 15);
    const float* W = (L == 0) ? w_1 : (L == 1) ? w_2 : (L == 2) ? w_3 : (L == 3) ? w_4 : (L == 4) ? hw1 : hw2;
    float v = 0.f;
    if (K < Kin[L] && N < Nout[L]) v = W[N * Kin[L] + K];
    wB[idx] = (_Float16)v;
  }
}

// ---------- Kernel 1: h1 = x @ c1_Wx^T, plus attention dot precompute ----------
__global__ void k_node1(const float* __restrict__ x, const float* __restrict__ W,
                        const float* __restrict__ as, const float* __restrict__ ad,
                        float* __restrict__ h1, float* __restrict__ s1, float* __restrict__ d1) {
  int i = blockIdx.x * blockDim.x + threadIdx.x;
  if (i >= NN) return;
  float xi[4];
#pragma unroll
  for (int k = 0; k < 4; ++k) xi[k] = x[i * 4 + k];
  float sa = 0.f, sd = 0.f;
#pragma unroll
  for (int j = 0; j < 16; ++j) {
    float hv = 0.f;
#pragma unroll
    for (int k = 0; k < 4; ++k) hv += xi[k] * W[j * 4 + k];
    h1[i * 16 + j] = hv;
    sa += hv * as[j];
    sd += hv * ad[j];
  }
  s1[i] = sa; d1[i] = sd;
}

// ---------- Kernel 2: conv1 attention logits + segment max ----------
__global__ void k_alpha1(const int* __restrict__ ei, const float* __restrict__ ea,
                         const float* __restrict__ wv, const float* __restrict__ s1,
                         const float* __restrict__ d1, float* __restrict__ al,
                         unsigned* __restrict__ m1) {
  int e = blockIdx.x * blockDim.x + threadIdx.x;
  int s = ei[e], d = ei[NE + e];
  float a = s1[s] + d1[d];
#pragma unroll
  for (int k = 0; k < 6; ++k) a += ea[e * 6 + k] * wv[k];
  a = a > 0.f ? a : 0.2f * a;
  al[e] = a;
  atomicMax(&m1[d], enc_ord(a));
}

// ---------- exp(alpha - max) + segment sum (shared by both convs) ----------
__global__ void k_ex(const int* __restrict__ dst, const unsigned* __restrict__ m,
                     float* __restrict__ al, float* __restrict__ sum) {
  int e = blockIdx.x * blockDim.x + threadIdx.x;
  int d = dst[e];
  float mm = dec_ord(m[d]);
  if (!(fabsf(mm) < 3.0e38f)) mm = 0.f;     // non-finite -> 0, like reference
  float ex = __expf(al[e] - mm);
  al[e] = ex;
  atomicAdd(&sum[d], ex);
}

// ---------- weighted message scatter (shared by both convs) ----------
__global__ void k_scat(const int* __restrict__ ei, const float* __restrict__ al,
                       const float* __restrict__ sum, const float* __restrict__ h,
                       float* __restrict__ acc) {
  int e = blockIdx.x * blockDim.x + threadIdx.x;
  int s = ei[e], d = ei[NE + e];
  float a = al[e] / (sum[d] + 1e-16f);
  const float4* hp = (const float4*)(h + s * 16);
  float4 q0 = hp[0], q1 = hp[1], q2 = hp[2], q3 = hp[3];
  float msg[16] = {q0.x,q0.y,q0.z,q0.w, q1.x,q1.y,q1.z,q1.w,
                   q2.x,q2.y,q2.z,q2.w, q3.x,q3.y,q3.z,q3.w};
  float* ap = acc + d * 16;
#pragma unroll
  for (int j = 0; j < 16; ++j) atomicAdd(&ap[j], a * msg[j]);
}

// ---------- Kernel 5: finalize x1 (+bias), compute h2 + conv2 attention dots ----------
__global__ void k_node2(float* __restrict__ x1, const float* __restrict__ b,
                        const float* __restrict__ W2, const float* __restrict__ as,
                        const float* __restrict__ ad, float* __restrict__ h2,
                        float* __restrict__ s2, float* __restrict__ d2) {
  int i = blockIdx.x * blockDim.x + threadIdx.x;
  if (i >= NN) return;
  float v[16];
#pragma unroll
  for (int j = 0; j < 16; ++j) { v[j] = x1[i * 16 + j] + b[j]; x1[i * 16 + j] = v[j]; }
  float sa = 0.f, sd = 0.f;
#pragma unroll
  for (int j = 0; j < 16; ++j) {
    float hv = 0.f;
#pragma unroll
    for (int k = 0; k < 16; ++k) hv += v[k] * W2[j * 16 + k];
    h2[i * 16 + j] = hv;
    sa += hv * as[j];
    sd += hv * ad[j];
  }
  s2[i] = sa; d2[i] = sd;
}

// ---------- Kernel 6: WMMA edge MLP + fused edge head + conv2 alpha (16 edges / wave) ----------
__global__ __launch_bounds__(128)
void k_mlp(const float* __restrict__ x1, const int* __restrict__ ei,
           const float* __restrict__ ea, const _Float16* __restrict__ wB,
           const float* __restrict__ w2v, const float* __restrict__ s2,
           const float* __restrict__ d2, const float* __restrict__ b1v,
           const float* __restrict__ b2v, const float* __restrict__ b3v,
           const float* __restrict__ b4v, const float* __restrict__ hb1,
           const float* __restrict__ hb2, float* __restrict__ alpha2,
           unsigned* __restrict__ m2, float* __restrict__ eout) {
  __shared__ __align__(16) float lds[4][2][16][LDW];
  int wvid = threadIdx.x >> 5;
  int lane = threadIdx.x & 31;
  int tile = blockIdx.x * 4 + wvid;
  int ebase = tile * 16;
  float (*bA)[LDW] = lds[wvid][0];
  float (*bB)[LDW] = lds[wvid][1];

  // stage input rows: [x1[src](0..15) | ea(16..21) | x1[dst](22..37) | 0(38..63)]
  if (lane < 16) {
    int e = ebase + lane;
    int s = ei[e];
    const float4* xp = (const float4*)(x1 + s * 16);
    float4 q0 = xp[0], q1 = xp[1], q2 = xp[2], q3 = xp[3];
    float* r = bA[lane];
    r[0]=q0.x; r[1]=q0.y; r[2]=q0.z; r[3]=q0.w;
    r[4]=q1.x; r[5]=q1.y; r[6]=q1.z; r[7]=q1.w;
    r[8]=q2.x; r[9]=q2.y; r[10]=q2.z; r[11]=q2.w;
    r[12]=q3.x; r[13]=q3.y; r[14]=q3.z; r[15]=q3.w;
#pragma unroll
    for (int k = 0; k < 6; ++k) r[16 + k] = ea[e * 6 + k];
  } else {
    int row = lane - 16;
    int e = ebase + row;
    int d = ei[NE + e];
    const float4* xp = (const float4*)(x1 + d * 16);
    float4 q0 = xp[0], q1 = xp[1], q2 = xp[2], q3 = xp[3];
    float* r = bA[row];
    r[22]=q0.x; r[23]=q0.y; r[24]=q0.z; r[25]=q0.w;
    r[26]=q1.x; r[27]=q1.y; r[28]=q1.z; r[29]=q1.w;
    r[30]=q2.x; r[31]=q2.y; r[32]=q2.z; r[33]=q2.w;
    r[34]=q3.x; r[35]=q3.y; r[36]=q3.z; r[37]=q3.w;
#pragma unroll
    for (int c = 38; c < 64; ++c) r[c] = 0.f;
  }
  WAVE_SYNC();

  // L1: 38(->64) -> 32, relu
  v16h a0 = loadA(bA, lane, 0);
  v16h a1 = loadA(bA, lane, 1);
#pragma unroll
  for (int nc = 0; nc < 2; ++nc) {
    v8f acc = {};
    acc = WMMA(a0, loadB(wB, 0 + nc, lane), acc);
    acc = WMMA(a1, loadB(wB, 2 + nc, lane), acc);
    storeD(bB, lane, acc, nc, b1v, true);
  }
  WAVE_SYNC();
  // L2: 32 -> 64, relu
  a0 = loadA(bB, lane, 0);
#pragma unroll
  for (int nc = 0; nc < 4; ++nc) {
    v8f acc = {};
    acc = WMMA(a0, loadB(wB, 4 + nc, lane), acc);
    storeD(bA, lane, acc, nc, b2v, true);
  }
  WAVE_SYNC();
  // L3: 64 -> 32, relu
  a0 = loadA(bA, lane, 0);
  a1 = loadA(bA, lane, 1);
#pragma unroll
  for (int nc = 0; nc < 2; ++nc) {
    v8f acc = {};
    acc = WMMA(a0, loadB(wB, 8 + nc, lane), acc);
    acc = WMMA(a1, loadB(wB, 10 + nc, lane), acc);
    storeD(bB, lane, acc, nc, b3v, true);
  }
  WAVE_SYNC();
  // L4: 32 -> 16 (= e), no relu; zero-pad cols 16..31
  a0 = loadA(bB, lane, 0);
  {
    v8f acc = {};
    acc = WMMA(a0, loadB(wB, 12, lane), acc);
    storeD(bA, lane, acc, 0, b4v, false);
    int zc = 16 + (lane & 15), rb = (lane >> 4) * 8;
#pragma unroll
    for (int r = 0; r < 8; ++r) bA[rb + r][zc] = 0.f;
  }
  WAVE_SYNC();
  // conv2 attention logit: alpha = lrelu(s2[src] + d2[dst] + e . (c2_We^T a_edge2))
  if (lane < 16) {
    int e = ebase + lane;
    int s = ei[e], d = ei[NE + e];
    float al = s2[s] + d2[d];
#pragma unroll
    for (int k = 0; k < 16; ++k) al += bA[lane][k] * w2v[k];
    al = al > 0.f ? al : 0.2f * al;
    alpha2[e] = al;
    atomicMax(&m2[d], enc_ord(al));
  }
  // edge head L1: 16 -> 16, relu; zero-pad cols 16..31
  a0 = loadA(bA, lane, 0);
  {
    v8f acc = {};
    acc = WMMA(a0, loadB(wB, 13, lane), acc);
    storeD(bB, lane, acc, 0, hb1, true);
    int zc = 16 + (lane & 15), rb = (lane >> 4) * 8;
#pragma unroll
    for (int r = 0; r < 8; ++r) bB[rb + r][zc] = 0.f;
  }
  WAVE_SYNC();
  // edge head L2: 16 -> 4, relu; stage to bA cols 0..3
  a0 = loadA(bB, lane, 0);
  {
    v8f acc = {};
    acc = WMMA(a0, loadB(wB, 14, lane), acc);
    int col = lane & 15, rb = (lane >> 4) * 8;
    if (col < 4) {
      float b = hb2[col];
#pragma unroll
      for (int r = 0; r < 8; ++r) bA[rb + r][col] = fmaxf(acc[r] + b, 0.f);
    }
  }
  WAVE_SYNC();
  // per-edge log_softmax over 4 classes -> output
  if (lane < 16) {
    float v0 = bA[lane][0], v1 = bA[lane][1], v2 = bA[lane][2], v3 = bA[lane][3];
    float mx = fmaxf(fmaxf(v0, v1), fmaxf(v2, v3));
    float ssum = __expf(v0 - mx) + __expf(v1 - mx) + __expf(v2 - mx) + __expf(v3 - mx);
    float ls = __logf(ssum);
    float* o = eout + (size_t)(ebase + lane) * 4;
    o[0] = v0 - mx - ls; o[1] = v1 - mx - ls; o[2] = v2 - mx - ls; o[3] = v3 - mx - ls;
  }
}

// ---------- Kernel 9: node head + log_softmax(2) ----------
__global__ void k_nodehead(const float* __restrict__ x2, const float* __restrict__ c2b,
                           const float* __restrict__ w1, const float* __restrict__ b1,
                           const float* __restrict__ w2, const float* __restrict__ b2,
                           float* __restrict__ out) {
  int i = blockIdx.x * blockDim.x + threadIdx.x;
  if (i >= NN) return;
  float v[16];
#pragma unroll
  for (int j = 0; j < 16; ++j) v[j] = x2[i * 16 + j] + c2b[j];
  float t[16];
#pragma unroll
  for (int j = 0; j < 16; ++j) {
    float s = b1[j];
#pragma unroll
    for (int k = 0; k < 16; ++k) s += v[k] * w1[j * 16 + k];
    t[j] = fmaxf(s, 0.f);
  }
  float o0 = b2[0], o1 = b2[1];
#pragma unroll
  for (int k = 0; k < 16; ++k) { o0 += t[k] * w2[k]; o1 += t[k] * w2[16 + k]; }
  o0 = fmaxf(o0, 0.f); o1 = fmaxf(o1, 0.f);
  float mx = fmaxf(o0, o1);
  float ls = __logf(__expf(o0 - mx) + __expf(o1 - mx));
  out[i * 2]     = o0 - mx - ls;
  out[i * 2 + 1] = o1 - mx - ls;
}

extern "C" void kernel_launch(void* const* d_in, const int* in_sizes, int n_in,
                              void* d_out, int out_size, void* d_ws, size_t ws_size,
                              hipStream_t stream) {
  const float* x     = (const float*)d_in[0];
  const int*   ei    = (const int*)d_in[1];
  const float* ea    = (const float*)d_in[2];
  const float* c1_Wx = (const float*)d_in[3];
  const float* c1_We = (const float*)d_in[4];
  const float* c1_as = (const float*)d_in[5];
  const float* c1_ad = (const float*)d_in[6];
  const float* c1_ae = (const float*)d_in[7];
  const float* c1_b  = (const float*)d_in[8];
  const float* e1w1  = (const float*)d_in[9];
  const float* e1b1  = (const float*)d_in[10];
  const float* e1w2  = (const float*)d_in[11];
  const float* e1b2  = (const float*)d_in[12];
  const float* e1w3  = (const float*)d_in[13];
  const float* e1b3  = (const float*)d_in[14];
  const float* e1w4  = (const float*)d_in[15];
  const float* e1b4  = (const float*)d_in[16];
  const float* c2_Wx = (const float*)d_in[17];
  const float* c2_We = (const float*)d_in[18];
  const float* c2_as = (const float*)d_in[19];
  const float* c2_ad = (const float*)d_in[20];
  const float* c2_ae = (const float*)d_in[21];
  const float* c2_b  = (const float*)d_in[22];
  const float* nlw1  = (const float*)d_in[23];
  const float* nlb1  = (const float*)d_in[24];
  const float* nlw2  = (const float*)d_in[25];
  const float* nlb2  = (const float*)d_in[26];
  const float* elw1  = (const float*)d_in[27];
  const float* elb1  = (const float*)d_in[28];
  const float* elw2  = (const float*)d_in[29];
  const float* elb2  = (const float*)d_in[30];
  float* out = (float*)d_out;
  float* ws  = (float*)d_ws;

  float*    h1   = ws;
  float*    s1   = ws + 1600000;
  float*    d1   = ws + 1700000;
  unsigned* m1   = (unsigned*)(ws + 1800000);
  float*    sum1 = ws + 1900000;
  float*    x1   = ws + 2000000;
  float*    h2   = ws + 3600000;
  float*    s2   = ws + 5200000;
  float*    d2   = ws + 5300000;
  unsigned* m2   = (unsigned*)(ws + 5400000);
  float*    sum2 = ws + 5500000;
  float*    x2   = ws + 5600000;
  float*    al1  = ws + 7200000;
  float*    al2  = ws + 10400000;
  float*    wvec = ws + 13600000;
  _Float16* wB   = (_Float16*)(ws + 13600032);   // 32B-aligned for v16h loads

  hipMemsetAsync(m1,   0, (size_t)NN * 4,  stream);
  hipMemsetAsync(sum1, 0, (size_t)NN * 4,  stream);
  hipMemsetAsync(x1,   0, (size_t)NN * 64, stream);
  hipMemsetAsync(m2,   0, (size_t)NN * 4,  stream);
  hipMemsetAsync(sum2, 0, (size_t)NN * 4,  stream);
  hipMemsetAsync(x2,   0, (size_t)NN * 64, stream);

  k_prep<<<1, 256, 0, stream>>>(c1_We, c1_ae, c2_We, c2_ae,
                                e1w1, e1w2, e1w3, e1w4, elw1, elw2, wvec, wB);
  k_node1<<<(NN + 255) / 256, 256, 0, stream>>>(x, c1_Wx, c1_as, c1_ad, h1, s1, d1);
  k_alpha1<<<NE / 256, 256, 0, stream>>>(ei, ea, wvec, s1, d1, al1, m1);
  k_ex<<<NE / 256, 256, 0, stream>>>(ei + NE, m1, al1, sum1);
  k_scat<<<NE / 256, 256, 0, stream>>>(ei, al1, sum1, h1, x1);
  k_node2<<<(NN + 255) / 256, 256, 0, stream>>>(x1, c1_b, c2_Wx, c2_as, c2_ad, h2, s2, d2);
  k_mlp<<<NE / 64, 128, 0, stream>>>(x1, ei, ea, wB, wvec + 8, s2, d2,
                                     e1b1, e1b2, e1b3, e1b4, elb1, elb2,
                                     al2, m2, out + (size_t)NN * 2);
  k_ex<<<NE / 256, 256, 0, stream>>>(ei + NE, m2, al2, sum2);
  k_scat<<<NE / 256, 256, 0, stream>>>(ei, al2, sum2, h2, x2);
  k_nodehead<<<(NN + 255) / 256, 256, 0, stream>>>(x2, c2_b, nlw1, nlb1, nlw2, nlb2, out);
}
